// EGNNLayer_17463337025565
// MI455X (gfx1250) — matrix operand order repaired and checked by reading, not compiled, not run on hardware
//
#include <hip/hip_runtime.h>
#include <hip/hip_bf16.h>

// EGNN layer, N=1024 fully-connected, for gfx1250 (MI455X, wave32, WMMA).
//
// Factorization: edge_in@ew1 + eb1 = (h_i@Wa + eb1) + (h_j@Wb) + r2*w_r
//   -> per-node A,B precompute kills the 129-K GEMM; the N^2 work is two
//      K<=128 f16 WMMA GEMMs per 16-edge tile plus elementwise silu.
// Round 2: fast-rcp silu (kill IEEE-div expansion), scalar-hoisted
// self-edge masking, packed f32 math + pk f16 converts in the U-build.
// Round 3: fix cvt_pkrtz return type (__fp16 vector, not _Float16 vector).

#define NN     1024
#define HDIM   64
#define MSGD   64
#define HIDD   128
#define SPLITS 4                    // j-splits == waves per block (grid.y==1)
#define JSPAN  (NN / SPLITS)        // 256 j per wave
#define NCHUNK (JSPAN / 16)         // 16 chunks of 16 edges

typedef __attribute__((ext_vector_type(16))) _Float16 v16h;
typedef __attribute__((ext_vector_type(2)))  __fp16   v2fp16;
typedef __attribute__((ext_vector_type(8)))  float    v8f;
typedef __attribute__((ext_vector_type(2)))  float    v2f;

// silu with single-instruction reciprocal (v_rcp_f32) instead of IEEE div.
__device__ __forceinline__ float silu(float v) {
    return v * __builtin_amdgcn_rcpf(1.0f + __expf(-v));
}

// A-operand (16x32 f16): lane l16 = row; halves = K { kb+8*hf+0..7, kb+16+8*hf+0..7 }
__device__ __forceinline__ v16h lds_loadA(const _Float16* base, int strideH,
                                          int row, int kbase, int hf) {
    union { v16h v; float4 f[2]; } u;
    const _Float16* p = base + row * strideH + kbase + 8 * hf;
    u.f[0] = *reinterpret_cast<const float4*>(p);
    u.f[1] = *reinterpret_cast<const float4*>(p + 16);
    return u.v;
}

// B-operand (32x16 f16), pre-swizzled: 16 contiguous halves per lane per block.
__device__ __forceinline__ v16h lds_loadB(const _Float16* blkbase, int L) {
    union { v16h v; float4 f[2]; } u;
    const _Float16* p = blkbase + L * 16;
    u.f[0] = *reinterpret_cast<const float4*>(p);
    u.f[1] = *reinterpret_cast<const float4*>(p + 8);
    return u.v;
}

// ---------------- Phase 0: A = h@Wa + eb1, B = h@Wb  (N x 128 each) --------
__global__ __launch_bounds__(128) void egnn_node_pre(
    const float* __restrict__ h, const float* __restrict__ ew1,
    const float* __restrict__ eb1, float* __restrict__ Amat,
    float* __restrict__ Bmat) {
    __shared__ float hs[HDIM];
    const int n = blockIdx.x, t = threadIdx.x;
    if (t < HDIM) hs[t] = h[n * HDIM + t];
    __syncthreads();
    float a = eb1[t], b = 0.0f;
    for (int k = 0; k < HDIM; ++k) {
        const float hv = hs[k];
        a = fmaf(hv, ew1[k * HIDD + t], a);
        b = fmaf(hv, ew1[(HDIM + k) * HIDD + t], b);
    }
    Amat[n * HIDD + t] = a;
    Bmat[n * HIDD + t] = b;
}

// ---------------- Phase 1: per-edge WMMA pipeline --------------------------
__global__ __launch_bounds__(128) void egnn_edge_kernel(
    const float* __restrict__ x,   const float* __restrict__ ew1,
    const float* __restrict__ ew2, const float* __restrict__ eb2,
    const float* __restrict__ cw1, const float* __restrict__ cb1,
    const float* __restrict__ cw2, const float* __restrict__ cb2,
    const float* __restrict__ Amat, const float* __restrict__ Bmat,
    float* __restrict__ mpart, float* __restrict__ xpart) {
    __shared__ __align__(16) _Float16 ew2s[HIDD * MSGD];   // 16 KB, B-swizzled
    __shared__ __align__(16) _Float16 cw1s[MSGD * HIDD];   // 16 KB, B-swizzled
    __shared__ __align__(16) _Float16 Us[4][16 * HIDD];    // 16 KB, per-wave U
    __shared__ __align__(16) _Float16 Ms[4][16 * MSGD];    //  8 KB, per-wave M
    __shared__ float r2s[4][16];

    const int tid = threadIdx.x;
    const int wv  = tid >> 5;
    const int L   = tid & 31;
    const int hf  = L >> 4;
    const int l16 = L & 15;
    const int i   = blockIdx.x;

    // Pre-swizzle weights into WMMA B-operand lane order (once per block).
    // ew2: K=128, N=64  -> 16 blocks (kk=0..3, nt=0..3) of 512 halves.
    for (int idx = tid; idx < HIDD * MSGD; idx += 128) {
        const int blk = idx >> 9, rem = idx & 511, lw = rem >> 4, hh = rem & 15;
        const int K = 32 * (blk >> 2) + 8 * (lw >> 4) + (hh & 7) + 16 * (hh >> 3);
        const int Nc = 16 * (blk & 3) + (lw & 15);
        ew2s[idx] = (_Float16)ew2[K * MSGD + Nc];
    }
    // cw1: K=64, N=128 -> 16 blocks (kk=0..1, nt=0..7).
    for (int idx = tid; idx < MSGD * HIDD; idx += 128) {
        const int blk = idx >> 9, rem = idx & 511, lw = rem >> 4, hh = rem & 15;
        const int K = 32 * (blk >> 3) + 8 * (lw >> 4) + (hh & 7) + 16 * (hh >> 3);
        const int Nc = 16 * (blk & 7) + (lw & 15);
        cw1s[idx] = (_Float16)cw1[K * HIDD + Nc];
    }
    __syncthreads();

    // Per-lane constants: lane covers k = 4L..4L+3 of the 128-hidden dim.
    const float4* Bmat4 = reinterpret_cast<const float4*>(Bmat);
    const float4 aieb = reinterpret_cast<const float4*>(Amat)[i * 32 + L];
    const float4 wr   = reinterpret_cast<const float4*>(ew1 + 128 * HIDD)[L];
    const v2f a01 = {aieb.x, aieb.y}, a23 = {aieb.z, aieb.w};
    const v2f w01 = {wr.x, wr.y},     w23 = {wr.z, wr.w};
    const float xi0 = x[i * 3 + 0], xi1 = x[i * 3 + 1], xi2 = x[i * 3 + 2];
    float eb2v[4], cb1v[8], cw2v[8];
#pragma unroll
    for (int nt = 0; nt < 4; ++nt) eb2v[nt] = eb2[16 * nt + l16];
#pragma unroll
    for (int nt = 0; nt < 8; ++nt) { cb1v[nt] = cb1[16 * nt + l16];
                                     cw2v[nt] = cw2[16 * nt + l16]; }
    const float cb2s = cb2[0];

    const int s = blockIdx.y * 4 + wv;
    const int j0base = s * JSPAN;
    float macc[4] = {0.f, 0.f, 0.f, 0.f};
    float xacc0 = 0.f, xacc1 = 0.f, xacc2 = 0.f;

    for (int c = 0; c < NCHUNK; ++c) {
        const int j0 = j0base + 16 * c;
        // dx / r2 for this lane's edge row jj = l16 (both halves compute it).
        const int jr = j0 + l16;
        const float dx0 = xi0 - x[jr * 3 + 0];
        const float dx1 = xi1 - x[jr * 3 + 1];
        const float dx2 = xi2 - x[jr * 3 + 2];
        const float r2v = dx0 * dx0 + dx1 * dx1 + dx2 * dx2;
        if (hf == 0) r2s[wv][l16] = r2v;
        if (c + 1 < NCHUNK)
            __builtin_prefetch(&Bmat4[(size_t)(j0 + 16) * 32 + L], 0, 1);

        // U[jj][k] = silu(A_i[k] + B_j[k] + r2*wr[k]) staged as f16 rows.
#pragma unroll 4
        for (int jj = 0; jj < 16; ++jj) {
            const float r2j = r2s[wv][jj];
            const float4 b = Bmat4[(size_t)(j0 + jj) * 32 + L];
            const v2f r2p = {r2j, r2j};
            const v2f t01 = a01 + (v2f){b.x, b.y} + r2p * w01;  // v_pk_fma/add
            const v2f t23 = a23 + (v2f){b.z, b.w} + r2p * w23;
            union { v2fp16 h2[2]; float2 f; } u;
            u.h2[0] = __builtin_amdgcn_cvt_pkrtz(silu(t01.x), silu(t01.y));
            u.h2[1] = __builtin_amdgcn_cvt_pkrtz(silu(t23.x), silu(t23.y));
            *reinterpret_cast<float2*>(&Us[wv][jj * HIDD + 4 * L]) = u.f;
        }

        // M(16x64) = U(16x128) @ ew2 + eb2, accumulators init with bias.
        v8f mcA[4];
#pragma unroll
        for (int nt = 0; nt < 4; ++nt) {
            v8f ci;
#pragma unroll
            for (int v = 0; v < 8; ++v) ci[v] = eb2v[nt];
            mcA[nt] = ci;
        }
#pragma unroll
        for (int kk = 0; kk < 4; ++kk) {
            const v16h a = lds_loadA(&Us[wv][0], HIDD, l16, 32 * kk, hf);
#pragma unroll
            for (int nt = 0; nt < 4; ++nt) {
                const v16h b = lds_loadB(&ew2s[(kk * 4 + nt) * 512], L);
                mcA[nt] = __builtin_amdgcn_wmma_f32_16x16x32_f16(
                    false, a, false, b, (short)0, mcA[nt], false, false);
            }
        }
        // Accumulate m_i, stage M as f16. Self-edge masking only runs in the
        // single chunk (wave-uniform test) whose j-range contains i.
        if ((i >> 4) == (j0 >> 4)) {
#pragma unroll
            for (int nt = 0; nt < 4; ++nt) {
#pragma unroll
                for (int v = 0; v < 8; ++v) {
                    const int row = v + 8 * hf;
                    float mv = mcA[nt][v];
                    if (j0 + row == i) mv = 0.f;
                    macc[nt] += mv;
                    Ms[wv][row * MSGD + 16 * nt + l16] = (_Float16)mv;
                }
            }
        } else {
#pragma unroll
            for (int nt = 0; nt < 4; ++nt) {
#pragma unroll
                for (int v = 0; v < 8; ++v) {
                    const int row = v + 8 * hf;
                    const float mv = mcA[nt][v];
                    macc[nt] += mv;
                    Ms[wv][row * MSGD + 16 * nt + l16] = (_Float16)mv;
                }
            }
        }

        // Gpre(16x128) = M(16x64) @ cw1 + cb1.
        v8f gc[8];
#pragma unroll
        for (int nt = 0; nt < 8; ++nt) {
            v8f ci;
#pragma unroll
            for (int v = 0; v < 8; ++v) ci[v] = cb1v[nt];
            gc[nt] = ci;
        }
#pragma unroll
        for (int kk = 0; kk < 2; ++kk) {
            const v16h a = lds_loadA(&Ms[wv][0], MSGD, l16, 32 * kk, hf);
#pragma unroll
            for (int nt = 0; nt < 8; ++nt) {
                const v16h b = lds_loadB(&cw1s[(kk * 8 + nt) * 512], L);
                gc[nt] = __builtin_amdgcn_wmma_f32_16x16x32_f16(
                    false, a, false, b, (short)0, gc[nt], false, false);
            }
        }

        // phi_e = cb2 + sum_k silu(g)*cw2 : lane partials + xor-reduce.
        float p[8];
#pragma unroll
        for (int v = 0; v < 8; ++v) p[v] = 0.f;
#pragma unroll
        for (int nt = 0; nt < 8; ++nt) {
            const float c2 = cw2v[nt];
#pragma unroll
            for (int v = 0; v < 8; ++v) p[v] = fmaf(silu(gc[nt][v]), c2, p[v]);
        }
#pragma unroll
        for (int v = 0; v < 8; ++v) {
            p[v] += __shfl_xor(p[v], 1);
            p[v] += __shfl_xor(p[v], 2);
            p[v] += __shfl_xor(p[v], 4);
            p[v] += __shfl_xor(p[v], 8);
            p[v] += cb2s;
        }
        // Lane whose l16 row lives in its own half picks phi(l16) and
        // accumulates dx*phi (each edge row counted exactly once).
        if ((l16 >> 3) == hf) {
            const int rsel = l16 & 7;
            float ph = p[0];
#pragma unroll
            for (int v = 1; v < 8; ++v) ph = (rsel == v) ? p[v] : ph;
            xacc0 = fmaf(dx0, ph, xacc0);
            xacc1 = fmaf(dx1, ph, xacc1);
            xacc2 = fmaf(dx2, ph, xacc2);
        }
    }

    // Write per-split partial m_i (fold the two lane halves first).
#pragma unroll
    for (int nt = 0; nt < 4; ++nt) {
        const float msum = macc[nt] + __shfl_xor(macc[nt], 16);
        if (hf == 0)
            mpart[(size_t)s * (NN * MSGD) + (size_t)i * MSGD + 16 * nt + l16] = msum;
    }
#pragma unroll
    for (int m = 1; m < 32; m <<= 1) {
        xacc0 += __shfl_xor(xacc0, m);
        xacc1 += __shfl_xor(xacc1, m);
        xacc2 += __shfl_xor(xacc2, m);
    }
    if (L == 0) {
        float* xp = &xpart[(size_t)s * (NN * 4) + (size_t)i * 4];
        xp[0] = xacc0; xp[1] = xacc1; xp[2] = xacc2;
    }
}

// ---------------- Phase 2: partial reduce + node MLP + outputs -------------
__global__ __launch_bounds__(128) void egnn_node_post(
    const float* __restrict__ x,   const float* __restrict__ h,
    const float* __restrict__ nw1, const float* __restrict__ nb1,
    const float* __restrict__ nw2, const float* __restrict__ nb2,
    const float* __restrict__ mpart, const float* __restrict__ xpart,
    float* __restrict__ out) {
    __shared__ float ins[HIDD];
    __shared__ float hid[HIDD];
    const int n = blockIdx.x, t = threadIdx.x;
    if (t < HDIM) {
        ins[t] = h[n * HDIM + t];
    } else {
        float m = 0.f;
        for (int s = 0; s < SPLITS; ++s)
            m += mpart[(size_t)s * (NN * MSGD) + (size_t)n * MSGD + (t - HDIM)];
        ins[t] = m;
    }
    __syncthreads();
    float acc = nb1[t];
    for (int k = 0; k < HIDD; ++k) acc = fmaf(ins[k], nw1[k * HIDD + t], acc);
    hid[t] = silu(acc);
    __syncthreads();
    if (t < HDIM) {
        float dh = nb2[t];
        for (int k = 0; k < HIDD; ++k) dh = fmaf(hid[k], nw2[k * HDIM + t], dh);
        out[NN * 3 + n * HDIM + t] = h[n * HDIM + t] + dh;
    }
    if (t < 3) {
        float xu = 0.f;
        for (int s = 0; s < SPLITS; ++s) xu += xpart[s * (NN * 4) + n * 4 + t];
        out[n * 3 + t] = x[n * 3 + t] + xu * (1.0f / (1023.0f + 1e-8f));
    }
}

extern "C" void kernel_launch(void* const* d_in, const int* in_sizes, int n_in,
                              void* d_out, int out_size, void* d_ws, size_t ws_size,
                              hipStream_t stream) {
    (void)in_sizes; (void)n_in; (void)out_size; (void)ws_size;
    const float* x   = (const float*)d_in[0];
    const float* h   = (const float*)d_in[1];
    const float* ew1 = (const float*)d_in[2];
    const float* eb1 = (const float*)d_in[3];
    const float* ew2 = (const float*)d_in[4];
    const float* eb2 = (const float*)d_in[5];
    const float* nw1 = (const float*)d_in[6];
    const float* nb1 = (const float*)d_in[7];
    const float* nw2 = (const float*)d_in[8];
    const float* nb2 = (const float*)d_in[9];
    const float* cw1 = (const float*)d_in[10];
    const float* cb1 = (const float*)d_in[11];
    const float* cw2 = (const float*)d_in[12];
    const float* cb2 = (const float*)d_in[13];

    float* ws    = (float*)d_ws;
    float* Amat  = ws;                            // 1024*128
    float* Bmat  = ws + NN * HIDD;                // 1024*128
    float* mpart = ws + 2 * NN * HIDD;            // SPLITS*1024*64
    float* xpart = mpart + SPLITS * NN * MSGD;    // SPLITS*1024*4

    egnn_node_pre<<<NN, 128, 0, stream>>>(h, ew1, eb1, Amat, Bmat);
    egnn_edge_kernel<<<dim3(NN, 1), 128, 0, stream>>>(
        x, ew1, ew2, eb2, cw1, cb1, cw2, cb2, Amat, Bmat, mpart, xpart);
    egnn_node_post<<<NN, 128, 0, stream>>>(x, h, nw1, nb1, nw2, nb2,
                                           mpart, xpart, (float*)d_out);
}
